// ToMe16_mlp_hd64_17154099380585
// MI455X (gfx1250) — matrix operand order repaired and checked by reading, not compiled, not run on hardware
//
#include <hip/hip_runtime.h>
#include <hip/hip_bf16.h>
#include <math.h>

typedef __attribute__((ext_vector_type(16))) __bf16 v16bf;
typedef __attribute__((ext_vector_type(8)))  float  v8f;

// ---------------------------------------------------------------------------
// WMMA fragment loaders (16x16x32 bf16; A and B share contiguous k = 16*(lane>>4) + element)
// ---------------------------------------------------------------------------

// A fragment: 16(M) x 32(K) bf16, row-major source with stride lda.
// lane L: row = L&15 ; kHalf = (L>>4)*16
// vgpr j : K = kHalf+2j, kHalf+2j+1
__device__ inline v16bf load_a_frag(const __bf16* A, int lda, int rowBase,
                                    int kBase, int lane) {
    int m  = rowBase + (lane & 15);
    int kh = (lane >> 4) * 16;
    const __bf16* rowp = A + (size_t)m * lda + kBase;
    v16bf f;
#pragma unroll
    for (int j = 0; j < 8; ++j) {
        int kb = kh + 2 * j;
        f[2 * j]     = rowp[kb];
        f[2 * j + 1] = rowp[kb + 1];
    }
    return f;
}

// B fragment: 32(K) x 16(N) bf16 from row-major B (stride ldb).
// lane L: n = L&15 ; lanes 0-15 hold K=0..15, lanes 16-31 hold K=16..31
__device__ inline v16bf load_b_frag(const __bf16* B, int ldb, int kBase,
                                    int colBase, int lane) {
    int n  = colBase + (lane & 15);
    int kh = (lane >> 4) * 16;
    v16bf f;
#pragma unroll
    for (int e = 0; e < 16; ++e)
        f[e] = B[(size_t)(kBase + kh + e) * ldb + n];
    return f;
}

// B fragment for scores = a @ b^T : logical B[k][n] = bMat[n][k], bMat row-major (T x 64)
__device__ inline v16bf load_bt_frag(const __bf16* bMat, int colBase, int kBase,
                                     int lane) {
    int n  = colBase + (lane & 15);
    int kh = (lane >> 4) * 16;
    const __bf16* rowp = bMat + (size_t)n * 64 + kBase + kh;
    v16bf f;
#pragma unroll
    for (int e = 0; e < 16; ++e)
        f[e] = rowp[e];
    return f;
}

// ---------------------------------------------------------------------------
// Metric: mean over 16 heads of 64-dim, L2-normalize, split even/odd, to bf16
// one wave (32 lanes) per token, 2 dims per lane
// ---------------------------------------------------------------------------
__global__ void metric_kernel(const float* __restrict__ x, int p,
                              __bf16* __restrict__ mEven,
                              __bf16* __restrict__ mOdd) {
    int wave = (int)((blockIdx.x * blockDim.x + threadIdx.x) >> 5);
    int lane = threadIdx.x & 31;
    if (wave >= p) return;
    const float* xt = x + (size_t)wave * 1024;
    float m0 = 0.f, m1 = 0.f;
#pragma unroll
    for (int h = 0; h < 16; ++h) {
        m0 += xt[h * 64 + lane];
        m1 += xt[h * 64 + lane + 32];
    }
    m0 *= (1.0f / 16.0f);
    m1 *= (1.0f / 16.0f);
    float ss = m0 * m0 + m1 * m1;
#pragma unroll
    for (int mask = 1; mask < 32; mask <<= 1)
        ss += __shfl_xor(ss, mask, 32);
    float inv = rsqrtf(fmaxf(ss, 1e-30f));
    __bf16* dst = (wave & 1) ? mOdd : mEven;
    dst += (size_t)(wave >> 1) * 64;
    dst[lane]      = (__bf16)(m0 * inv);
    dst[lane + 32] = (__bf16)(m1 * inv);
}

// ---------------------------------------------------------------------------
// Fused similarity GEMM + row max/argmax.
// One wave owns a 16-row tile; loops over all 16-col tiles with WMMA bf16,
// keeps running (max, argmax) per row in registers, butterfly-reduces over
// the 16 N-lanes at the end. Never materializes the score matrix.
// ---------------------------------------------------------------------------
__global__ void scores_argmax_kernel(const __bf16* __restrict__ aM,
                                     const __bf16* __restrict__ bM,
                                     float* __restrict__ nodeMax,
                                     int* __restrict__ nodeIdx, int half) {
    int wave = (int)((blockIdx.x * blockDim.x + threadIdx.x) >> 5);
    int lane = threadIdx.x & 31;
    int rowBase = wave << 4;
    if (rowBase >= half) return;

    float bestV[8];
    int   bestI[8];
#pragma unroll
    for (int e = 0; e < 8; ++e) { bestV[e] = -INFINITY; bestI[e] = 0; }

    for (int cb = 0; cb < half; cb += 16) {
        // prefetch next column tile of bM (global_prefetch_b8)
        if (cb + 16 < half)
            __builtin_prefetch(bM + (size_t)(cb + 16 + (lane & 15)) * 64, 0, 1);
        v8f acc = {};
#pragma unroll
        for (int k = 0; k < 64; k += 32) {
            v16bf a = load_a_frag(aM, 64, rowBase, k, lane);
            v16bf b = load_bt_frag(bM, cb, k, lane);
            acc = __builtin_amdgcn_wmma_f32_16x16x32_bf16(
                false, a, false, b, (short)0, acc, false, false);
        }
        int col = cb + (lane & 15);
#pragma unroll
        for (int e = 0; e < 8; ++e) {
            float v = acc[e];
            if (v > bestV[e]) { bestV[e] = v; bestI[e] = col; }
        }
    }
    // reduce across the 16 N-lanes (masks 1..8 stay within 16-lane halves)
#pragma unroll
    for (int mask = 1; mask <= 8; mask <<= 1) {
#pragma unroll
        for (int e = 0; e < 8; ++e) {
            float ov = __shfl_xor(bestV[e], mask, 32);
            int   oi = __shfl_xor(bestI[e], mask, 32);
            if (ov > bestV[e] || (ov == bestV[e] && oi < bestI[e])) {
                bestV[e] = ov; bestI[e] = oi;
            }
        }
    }
    int mh = (lane >> 4) * 8;
    if ((lane & 15) == 0) {
#pragma unroll
        for (int e = 0; e < 8; ++e) {
            nodeMax[rowBase + mh + e] = bestV[e];
            nodeIdx[rowBase + mh + e] = bestI[e];
        }
    }
}

// ---------------------------------------------------------------------------
// Single-block bitonic argsort, descending by key, ties -> ascending index
// (matches stable argsort of -node_max). Dynamic LDS: P2*(4+4) bytes <=128KB
// (within the 320KB/WGP LDS budget of CDNA5).
// ---------------------------------------------------------------------------
__global__ void bitonic_argsort_kernel(const float* __restrict__ nodeMax,
                                       int* __restrict__ edgeIdx, int T, int P2) {
    extern __shared__ unsigned char smemraw[];
    float* key = (float*)smemraw;
    int*   val = (int*)(smemraw + (size_t)P2 * sizeof(float));
    for (int i = threadIdx.x; i < P2; i += blockDim.x) {
        if (i < T) { key[i] = nodeMax[i]; val[i] = i; }
        else       { key[i] = -INFINITY;  val[i] = 0x7fffffff; }
    }
    __syncthreads();
    for (int k = 2; k <= P2; k <<= 1) {
        for (int j = k >> 1; j > 0; j >>= 1) {
            for (int i = threadIdx.x; i < P2; i += blockDim.x) {
                int ij = i ^ j;
                if (ij > i) {
                    bool asc = ((i & k) == 0);
                    float ki = key[i], kj = key[ij];
                    int   vi = val[i], vj = val[ij];
                    bool iFirst = (ki > kj) || (ki == kj && vi < vj);
                    bool doSwap = asc ? !iFirst : iFirst;
                    if (doSwap) {
                        key[i] = kj; key[ij] = ki;
                        val[i] = vj; val[ij] = vi;
                    }
                }
            }
            __syncthreads();
        }
    }
    for (int i = threadIdx.x; i < T; i += blockDim.x)
        edgeIdx[i] = val[i];
}

// ---------------------------------------------------------------------------
// Merge kernels: newX = concat([unm (u tokens), dst (half tokens)]) of x*size;
// newSize likewise; then divide.
// ---------------------------------------------------------------------------
__global__ void merge_dst_init_kernel(const float* __restrict__ x,
                                      const float* __restrict__ sz,
                                      float* __restrict__ nx,
                                      float* __restrict__ nsz, int half, int u) {
    int idx = (int)(blockIdx.x * blockDim.x + threadIdx.x);
    int total = half * 1024;
    if (idx >= total) return;
    int j = idx >> 10, c = idx & 1023;
    float s = sz ? sz[2 * j + 1] : 1.0f;
    nx[(size_t)(u + j) * 1024 + c] = x[(size_t)(2 * j + 1) * 1024 + c] * s;
    if (c == 0) nsz[u + j] = s;
}

__global__ void merge_scatter_kernel(const float* __restrict__ x,
                                     const float* __restrict__ sz,
                                     const int* __restrict__ edge,
                                     const int* __restrict__ nodeIdx,
                                     float* __restrict__ nx,
                                     float* __restrict__ nsz, int r, int u) {
    int idx = (int)(blockIdx.x * blockDim.x + threadIdx.x);
    int total = r * 1024;
    if (idx >= total) return;
    int i = idx >> 10, c = idx & 1023;
    int s = edge[i];             // src_idx
    int d = nodeIdx[s];          // dst_idx
    float sv = sz ? sz[2 * s] : 1.0f;
    atomicAdd(&nx[(size_t)(u + d) * 1024 + c],
              x[(size_t)(2 * s) * 1024 + c] * sv);
    if (c == 0) atomicAdd(&nsz[u + d], sv);
}

__global__ void merge_unm_kernel(const float* __restrict__ x,
                                 const float* __restrict__ sz,
                                 const int* __restrict__ edge,
                                 float* __restrict__ nx,
                                 float* __restrict__ nsz, int r, int u) {
    int idx = (int)(blockIdx.x * blockDim.x + threadIdx.x);
    int total = u * 1024;
    if (idx >= total) return;
    int i = idx >> 10, c = idx & 1023;
    int s = edge[r + i];
    float sv = sz ? sz[2 * s] : 1.0f;
    nx[(size_t)i * 1024 + c] = x[(size_t)(2 * s) * 1024 + c] * sv;
    if (c == 0) nsz[i] = sv;
}

__global__ void merge_div_kernel(float* __restrict__ nx,
                                 const float* __restrict__ nsz, int newP) {
    int idx = (int)(blockIdx.x * blockDim.x + threadIdx.x);
    int total = newP * 1024;
    if (idx >= total) return;
    nx[idx] /= nsz[idx >> 10];
}

// ---------------------------------------------------------------------------
// f32 -> bf16 conversion
// ---------------------------------------------------------------------------
__global__ void cvt_bf16_kernel(const float* __restrict__ src,
                                __bf16* __restrict__ dst, long long n) {
    long long i = (long long)blockIdx.x * blockDim.x + threadIdx.x;
    if (i >= n) return;
    dst[i] = (__bf16)src[i];
}

// ---------------------------------------------------------------------------
// GEMM C = A(MxK) @ B(KxN) + bias [, exact GELU], WMMA bf16->f32.
// One wave computes a 16(M) x 64(N) tile: 4 accumulators share one A
// fragment per 32-wide k-step (4 WMMAs per A load; the 4 adjacent B
// sub-tiles give 128B-contiguous per-k-row access across the half-wave).
// ---------------------------------------------------------------------------
template <int GELU, int OUTBF>
__global__ void gemm_wmma_kernel(const __bf16* __restrict__ A,
                                 const __bf16* __restrict__ B,
                                 const float* __restrict__ bias, void* Cout,
                                 int M, int N, int K) {
    int wave = (int)((blockIdx.x * blockDim.x + threadIdx.x) >> 5);
    int lane = threadIdx.x & 31;
    int tilesN = N >> 6;                 // 64-wide N tiles
    int tiles  = (M >> 4) * tilesN;
    if (wave >= tiles) return;
    int tm = (wave / tilesN) << 4;
    int tn = (wave % tilesN) << 6;

    v8f acc0 = {}, acc1 = {}, acc2 = {}, acc3 = {};
    for (int k = 0; k < K; k += 32) {
        if (k + 32 < K) {
            // prefetch the next k-step's B rows (global_prefetch_b8)
            int kh = (lane >> 4) * 16;
            __builtin_prefetch(B + (size_t)(k + 32 + kh) * N + tn + (lane & 15) * 4, 0, 1);
        }
        v16bf a  = load_a_frag(A, K, tm, k, lane);
        v16bf b0 = load_b_frag(B, N, k, tn, lane);
        v16bf b1 = load_b_frag(B, N, k, tn + 16, lane);
        v16bf b2 = load_b_frag(B, N, k, tn + 32, lane);
        v16bf b3 = load_b_frag(B, N, k, tn + 48, lane);
        acc0 = __builtin_amdgcn_wmma_f32_16x16x32_bf16(false, a, false, b0,
                                                       (short)0, acc0, false, false);
        acc1 = __builtin_amdgcn_wmma_f32_16x16x32_bf16(false, a, false, b1,
                                                       (short)0, acc1, false, false);
        acc2 = __builtin_amdgcn_wmma_f32_16x16x32_bf16(false, a, false, b2,
                                                       (short)0, acc2, false, false);
        acc3 = __builtin_amdgcn_wmma_f32_16x16x32_bf16(false, a, false, b3,
                                                       (short)0, acc3, false, false);
    }

    int mh = (lane >> 4) * 8;
    v8f accs[4] = {acc0, acc1, acc2, acc3};
#pragma unroll
    for (int t = 0; t < 4; ++t) {
        int n = tn + t * 16 + (lane & 15);
        float bv = bias[n];
#pragma unroll
        for (int e = 0; e < 8; ++e) {
            int m = tm + mh + e;
            float v = accs[t][e] + bv;
            if (GELU) v = 0.5f * v * (1.0f + erff(v * 0.70710678118654752f));
            if (OUTBF)
                ((__bf16*)Cout)[(size_t)m * N + n] = (__bf16)v;
            else
                ((float*)Cout)[(size_t)m * N + n] = v;
        }
    }
}

// ---------------------------------------------------------------------------
// Host-side driver. Merge schedule is fully determined by shapes:
// p: 18432 -> 9216 -> 4608 -> 2304 -> 1152 -> 576 -> 512
// ---------------------------------------------------------------------------
extern "C" void kernel_launch(void* const* d_in, const int* in_sizes, int n_in,
                              void* d_out, int out_size, void* d_ws,
                              size_t ws_size, hipStream_t stream) {
    const float* x_in = (const float*)d_in[0];
    const float* W1   = (const float*)d_in[1];
    const float* b1   = (const float*)d_in[2];
    const float* W2   = (const float*)d_in[3];
    const float* b2   = (const float*)d_in[4];
    (void)in_sizes; (void)n_in; (void)out_size; (void)ws_size;

    const int P0 = 32 * 576;        // 18432 tokens after (1,-1,1024) reshape
    const int TARGET = 16 * 32;     // NUM_COMPRESSED_TOKENS * num_frames
    const int MAXHALF = P0 / 2;     // 9216

    // --- carve workspace ---
    unsigned char* ws = (unsigned char*)d_ws;
    size_t off = 0;
    auto carve = [&](size_t bytes) -> void* {
        void* p = ws + off;
        off += (bytes + 255) & ~(size_t)255;
        return p;
    };
    float*  bx0     = (float*)carve((size_t)MAXHALF * 1024 * sizeof(float));
    float*  bx1     = (float*)carve((size_t)MAXHALF * 1024 * sizeof(float));
    float*  bs0     = (float*)carve((size_t)MAXHALF * sizeof(float));
    float*  bs1     = (float*)carve((size_t)MAXHALF * sizeof(float));
    __bf16* mEven   = (__bf16*)carve((size_t)MAXHALF * 64 * sizeof(__bf16));
    __bf16* mOdd    = (__bf16*)carve((size_t)MAXHALF * 64 * sizeof(__bf16));
    float*  nodeMax = (float*)carve((size_t)MAXHALF * sizeof(float));
    int*    nodeIdx = (int*)carve((size_t)MAXHALF * sizeof(int));
    int*    edgeIdx = (int*)carve((size_t)MAXHALF * sizeof(int));
    __bf16* xbf     = (__bf16*)carve((size_t)TARGET * 1024 * sizeof(__bf16));
    __bf16* w1bf    = (__bf16*)carve((size_t)1024 * 4096 * sizeof(__bf16));
    __bf16* w2bf    = (__bf16*)carve((size_t)4096 * 4096 * sizeof(__bf16));
    __bf16* hbf     = (__bf16*)carve((size_t)TARGET * 4096 * sizeof(__bf16));

    // --- ToMe merge loop ---
    const float* xCur  = x_in;
    const float* szCur = nullptr;
    int sel = 0;
    int p = P0;
    while (p != TARGET) {
        int half = p / 2;
        int r = (p - TARGET <= half) ? (p - TARGET) : half;
        int u = half - r;
        int newP = half + u;

        metric_kernel<<<(p + 7) / 8, 256, 0, stream>>>(xCur, p, mEven, mOdd);

        int rowWaves = half / 16;
        scores_argmax_kernel<<<(rowWaves + 7) / 8, 256, 0, stream>>>(
            mEven, mOdd, nodeMax, nodeIdx, half);

        int P2 = 1;
        while (P2 < half) P2 <<= 1;
        bitonic_argsort_kernel<<<1, 1024, (size_t)P2 * 8, stream>>>(
            nodeMax, edgeIdx, half, P2);

        float* nx  = sel ? bx1 : bx0;
        float* nsz = sel ? bs1 : bs0;

        int t1 = half * 1024;
        merge_dst_init_kernel<<<(t1 + 255) / 256, 256, 0, stream>>>(
            xCur, szCur, nx, nsz, half, u);
        int t2 = r * 1024;
        merge_scatter_kernel<<<(t2 + 255) / 256, 256, 0, stream>>>(
            xCur, szCur, edgeIdx, nodeIdx, nx, nsz, r, u);
        if (u > 0) {
            int t3 = u * 1024;
            merge_unm_kernel<<<(t3 + 255) / 256, 256, 0, stream>>>(
                xCur, szCur, edgeIdx, nx, nsz, r, u);
        }
        int t4 = newP * 1024;
        merge_div_kernel<<<(t4 + 255) / 256, 256, 0, stream>>>(nx, nsz, newP);

        xCur = nx; szCur = nsz; sel ^= 1; p = newP;
    }

    // --- MLP: gelu(x @ W1 + b1) @ W2 + b2 ---
    {
        long long n;
        n = (long long)TARGET * 1024;
        cvt_bf16_kernel<<<(unsigned)((n + 255) / 256), 256, 0, stream>>>(xCur, xbf, n);
        n = (long long)1024 * 4096;
        cvt_bf16_kernel<<<(unsigned)((n + 255) / 256), 256, 0, stream>>>(W1, w1bf, n);
        n = (long long)4096 * 4096;
        cvt_bf16_kernel<<<(unsigned)((n + 255) / 256), 256, 0, stream>>>(W2, w2bf, n);
    }
    {
        int tiles = (TARGET / 16) * (4096 / 64);   // 16x64 tile per wave
        gemm_wmma_kernel<1, 1><<<(tiles + 7) / 8, 256, 0, stream>>>(
            xbf, w1bf, b1, (void*)hbf, TARGET, 4096, 1024);
        gemm_wmma_kernel<0, 0><<<(tiles + 7) / 8, 256, 0, stream>>>(
            hbf, w2bf, b2, d_out, TARGET, 4096, 4096);
    }
}
